// SoftPool_13812614824491
// MI455X (gfx1250) — compile-verified, matrix operand order checked by reading, not run on hardware
//
#include <hip/hip_runtime.h>
#include <hip/hip_bf16.h>

// ---------------------------------------------------------------------------
// SoftPool on MI455X (gfx1250, wave32)
//   B=8, C=256, N=8192, R=16, P=2048, CABINS=8
// Outputs (concatenated float): sp_cube[8,256,16,2048], sp_idx[8,19,16,2048],
//                               cabins[8,256,16,8],     id_activa[8,8192]
// ---------------------------------------------------------------------------

#define B_   8
#define C_   256
#define N_   8192
#define R_   16
#define P_   2048
#define CAB_ 8

typedef float  v2f  __attribute__((ext_vector_type(2)));
typedef float  v8f  __attribute__((ext_vector_type(8)));
typedef __bf16 v16bf __attribute__((ext_vector_type(16)));
typedef int    v4i  __attribute__((vector_size(4 * sizeof(int))));

#define GLOBAL_AS __attribute__((address_space(1)))
#define LDS_AS    __attribute__((address_space(3)))

#if defined(__has_builtin)
#if __has_builtin(__builtin_amdgcn_wmma_f32_16x16x4_f32)
#define HAVE_WMMA_F32X4 1
#endif
#if __has_builtin(__builtin_amdgcn_global_load_async_to_lds_b128) && \
    __has_builtin(__builtin_amdgcn_s_wait_asynccnt)
#define HAVE_ASYNC_LDS 1
#endif
#endif

// ---------------------------------------------------------------------------
// Kernel 1: sorter 1x1-conv via WMMA  +  argmax over regions.
//   grid = (64, 8) blocks of 256 threads (8 waves). Each wave owns 16 points:
//   D[16 regions x 16 points] = W[16x256] * X[256x16]  as 64 chained
//   v_wmma_f32_16x16x4_f32 (fp32 A/B, fp32 acc) — or bf16 16x16x32 fallback.
// Writes val_activa -> ws_val[b][r][n] (for top-k) and id_activa -> d_out.
// ---------------------------------------------------------------------------
__global__ __launch_bounds__(256) void sorter_wmma_kernel(
    const float* __restrict__ x,      // [B, C, N]
    const float* __restrict__ W,      // [R, C]
    const float* __restrict__ bias,   // [R]
    float* __restrict__ ws_val,       // [B, R, N]
    float* __restrict__ out_id)       // [B, N]   (argmax stored as float)
{
    __shared__ float lds_val[8 * 16 * 16];   // [wave][point 0..15][region 0..15]

    const int tid  = threadIdx.x;
    const int wave = tid >> 5;
    const int lane = tid & 31;
    const int half = (lane >= 16) ? 1 : 0;   // lane group (K/M split per ISA layout)
    const int ml   = lane & 15;              // A row (region) / B col (point)

    const int b  = blockIdx.y;
    const int n0 = blockIdx.x * 128 + wave * 16;   // 16 points per wave
    const int n  = n0 + ml;                        // this lane's point (B/C/D col)

    const float* xb = x + ((size_t)b * C_) * N_;

    v8f acc = {};

#if HAVE_WMMA_F32X4
    // f32 A 16x4 layout: VGPR0 = K (lanes 0-15) / K+2 (lanes 16-31); VGPR1 = K+1 / K+3
    for (int k0 = 0; k0 < C_; k0 += 4) {
        const int ka = k0 + 2 * half;
        v2f a, bv;
        a.x  = W[ml * C_ + ka + 0];
        a.y  = W[ml * C_ + ka + 1];
        bv.x = xb[(size_t)(ka + 0) * N_ + n];
        bv.y = xb[(size_t)(ka + 1) * N_ + n];
        acc = __builtin_amdgcn_wmma_f32_16x16x4_f32(
            false, a, false, bv, (short)0, acc, false, false);
    }
#else
    // bf16 fallback (codegen-confirmed): 16-bit A 16x32 layout —
    // element e holds K = (e<8 ? e : e+8) + (lane>=16 ? 8 : 0) within the k0 tile.
    for (int k0 = 0; k0 < C_; k0 += 32) {
        v16bf a, bv;
        #pragma unroll
        for (int e = 0; e < 16; ++e) {
            const int k = k0 + (e < 8 ? e : e + 8) + (half ? 8 : 0);
            a[e]  = (__bf16)W[ml * C_ + k];
            bv[e] = (__bf16)xb[(size_t)k * N_ + n];
        }
        acc = __builtin_amdgcn_wmma_f32_16x16x32_bf16(
            false, a, false, bv, (short)0, acc, false, false);
    }
#endif

    // D layout: VGPR v holds row M = v + 8*half, col N = lane&15.
    #pragma unroll
    for (int v = 0; v < 8; ++v) {
        const int reg = v + 8 * half;
        const float val = acc[v] + bias[reg];
        ws_val[((size_t)b * R_ + reg) * N_ + n] = val;
        lds_val[wave * 256 + ml * 16 + reg] = val;
    }
    __syncthreads();

    // argmax over 16 regions: one thread per point (128 points per block)
    if (tid < 128) {
        const int pw = tid >> 4;          // wave that produced this point
        const int pl = tid & 15;          // point within wave
        const float* row = &lds_val[pw * 256 + pl * 16];
        float best = row[0];
        int   bi   = 0;
        #pragma unroll
        for (int r = 1; r < R_; ++r) {
            const float vr = row[r];
            if (vr > best) { best = vr; bi = r; }   // first-max tie break
        }
        out_id[(size_t)b * N_ + blockIdx.x * 128 + pw * 16 + pl] = (float)bi;
    }
}

// ---------------------------------------------------------------------------
// Kernel 2: per-(b,r) descending top-2048-of-8192 via in-LDS bitonic sort of
// packed 64-bit keys: (monotone f32 << 32) | ~index  (ties -> lower index).
// 64 KB LDS (320 KB/WGP budget), 1024 threads = 32 wave32 waves, 128 blocks.
// Writes idx -> ws_idx and the sp_idx broadcast (19 channel copies) -> d_out.
// ---------------------------------------------------------------------------
__global__ __launch_bounds__(1024) void topk_bitonic_kernel(
    const float* __restrict__ ws_val,    // [B, R, N]
    unsigned* __restrict__ ws_idx,       // [B, R, P]
    float* __restrict__ out_spidx)       // [B, 19, R, P]
{
    __shared__ unsigned long long keys[N_];   // 64 KB

    const int r = blockIdx.x;
    const int b = blockIdx.y;
    const float* v = ws_val + ((size_t)b * R_ + r) * N_;

    for (int i = threadIdx.x; i < N_; i += 1024) {
        unsigned u = __float_as_uint(v[i]);
        u = (u & 0x80000000u) ? ~u : (u | 0x80000000u);     // order-preserving map
        keys[i] = ((unsigned long long)u << 32) | (unsigned)(~i);
    }
    __syncthreads();

    for (int k = 2; k <= N_; k <<= 1) {
        for (int j = k >> 1; j > 0; j >>= 1) {
            for (int i = threadIdx.x; i < N_; i += 1024) {
                const int ixj = i ^ j;
                if (ixj > i) {
                    const unsigned long long a = keys[i];
                    const unsigned long long c = keys[ixj];
                    const bool desc = ((i & k) == 0);
                    if (desc ? (a < c) : (a > c)) { keys[i] = c; keys[ixj] = a; }
                }
            }
            __syncthreads();
        }
    }

    // keys[0..P-1] now hold the top-P in descending order.
    for (int p = threadIdx.x; p < P_; p += 1024) {
        const unsigned idx = ~(unsigned)(keys[p] & 0xFFFFFFFFu);
        ws_idx[((size_t)b * R_ + r) * P_ + p] = idx;
        const float fi = (float)idx;
        #pragma unroll
        for (int ch = 0; ch < R_ + 3; ++ch)
            out_spidx[(((size_t)b * (R_ + 3) + ch) * R_ + r) * P_ + p] = fi;
    }
}

// ---------------------------------------------------------------------------
// Kernel 3: gather sp_cube rows + cabin maxes.  grid = (c=256, r=16, b=8),
// 256 threads. The per-(b,r) index row (8 KB) is staged into LDS once per
// block with GLOBAL_LOAD_ASYNC_TO_LDS_B128 (ASYNCcnt / s_wait_asynccnt path);
// sp_cube rows are contiguous coalesced 8 KB writes; gathered x reads hit the
// 192 MB L2 (x is only 64 MB). Cabin maxes via LDS + wave32 shuffles.
// ---------------------------------------------------------------------------
__global__ __launch_bounds__(256) void gather_cabins_kernel(
    const float* __restrict__ x,         // [B, C, N]
    const unsigned* __restrict__ ws_idx, // [B, R, P]
    float* __restrict__ out_cube,        // [B, C, R, P]
    float* __restrict__ out_cab)         // [B, C, R, 8]
{
    __shared__ unsigned sidx[P_];        // 8 KB staged index row
    __shared__ float    red[P_];         // 8 KB cabin reduction buffer

    const int c = blockIdx.x;
    const int r = blockIdx.y;
    const int b = blockIdx.z;
    const int t = threadIdx.x;

    const unsigned* irow = ws_idx + ((size_t)b * R_ + r) * P_;
    const float*    xrow = x + ((size_t)b * C_ + c) * N_;
    float*          crow = out_cube + (((size_t)b * C_ + c) * R_ + r) * P_;

#if HAVE_ASYNC_LDS
    // Stage idx row: 2048 u32 = 8 KB as 512 async b128 copies (2 per thread).
    // Builtin signature (per clang diagnostic): (v4i __device__* src, ...)
    #pragma unroll
    for (int i = 0; i < 2; ++i) {
        const int e = (i * 256 + t) * 4;            // 4 u32 per b128
        __builtin_amdgcn_global_load_async_to_lds_b128(
            (GLOBAL_AS v4i*)(irow + e),
            (LDS_AS v4i*)&sidx[e],
            0, 0);
    }
    __builtin_amdgcn_s_wait_asynccnt(0);
    __syncthreads();
#else
    #pragma unroll
    for (int i = 0; i < CAB_; ++i) {
        const int p = i * 256 + t;
        sidx[p] = irow[p];
    }
    __syncthreads();
#endif

    #pragma unroll
    for (int i = 0; i < CAB_; ++i) {
        const int p = i * 256 + t;                  // element of cabin chunk i
        const float v = xrow[sidx[p]];
        crow[p] = v;
        red[p]  = v;
    }
    __syncthreads();

    // wave w (8 waves) reduces cabin chunk w (256 contiguous values)
    const int w    = t >> 5;
    const int lane = t & 31;
    float m = -__builtin_inff();
    for (int q = lane; q < 256; q += 32) m = fmaxf(m, red[w * 256 + q]);
    #pragma unroll
    for (int off = 16; off >= 1; off >>= 1)         // wave32: widths 16..1
        m = fmaxf(m, __shfl_xor(m, off, 32));
    if (lane == 0)
        out_cab[(((size_t)b * C_ + c) * R_ + r) * CAB_ + w] = m;
}

// ---------------------------------------------------------------------------
extern "C" void kernel_launch(void* const* d_in, const int* in_sizes, int n_in,
                              void* d_out, int out_size, void* d_ws, size_t ws_size,
                              hipStream_t stream) {
    const float* x    = (const float*)d_in[0];   // [8, 256, 8192]
    const float* W    = (const float*)d_in[1];   // [16, 256]
    const float* bias = (const float*)d_in[2];   // [16]

    float* out = (float*)d_out;
    const size_t n_cube  = (size_t)B_ * C_ * R_ * P_;        // 67,108,864
    const size_t n_spidx = (size_t)B_ * (R_ + 3) * R_ * P_;  //  4,980,736
    const size_t n_cab   = (size_t)B_ * C_ * R_ * CAB_;      //    262,144

    float* out_cube  = out;
    float* out_spidx = out + n_cube;
    float* out_cab   = out_spidx + n_spidx;
    float* out_id    = out_cab + n_cab;

    float*    ws_val = (float*)d_ws;                         // 4 MB: [B,R,N]
    unsigned* ws_idx = (unsigned*)(ws_val + (size_t)B_ * R_ * N_); // 1 MB

    sorter_wmma_kernel<<<dim3(N_ / 128, B_), 256, 0, stream>>>(
        x, W, bias, ws_val, out_id);
    topk_bitonic_kernel<<<dim3(R_, B_), 1024, 0, stream>>>(
        ws_val, ws_idx, out_spidx);
    gather_cabins_kernel<<<dim3(C_, R_, B_), 256, 0, stream>>>(
        x, ws_idx, out_cube, out_cab);
}
